// Network_10385230921885
// MI455X (gfx1250) — matrix operand order repaired and checked by reading, not compiled, not run on hardware
//
#include <hip/hip_runtime.h>
#include <hip/hip_bf16.h>

// ---------------------------------------------------------------------------
// SLAYER SNN forward, restructured for MI455X (gfx1250):
//   psp (alpha FIR, K=100)  -> exact 2-state IIR, fused with spike scan
//   linear ops commuted in front of psp (pool first at full res!)
//   fc1/fc2 -> f16 WMMA (v_wmma_f32_16x16x32_f16), f32 accumulate
// ---------------------------------------------------------------------------

#define THETA   10.0f
#define IIR_D   0.90483741803595952f   // exp(-Ts/tau_sr) = exp(-0.1)
#define IIR_EC  0.27182818284590454f   // e/tau_sr
#define REF_D   0.36787944117144233f   // exp(-Ts/tau_ref) = exp(-1)

typedef __attribute__((ext_vector_type(16))) _Float16 v16h;
typedef __attribute__((ext_vector_type(8)))  float    v8f;

static __host__ __device__ inline size_t cdiv(size_t a, size_t b) { return (a + b - 1) / b; }

// ---------------------------------------------------------------------------
// k x k sum-pool with SLAYER scale, per timestep.  in [BC,H,W,T] -> out [BC,Ho,Wo,T]
// ---------------------------------------------------------------------------
__global__ __launch_bounds__(256)
void pool_kernel(const float* __restrict__ in, float* __restrict__ out,
                 int BC, int H, int W, int k, int T, float scale)
{
    const int Ho = H / k, Wo = W / k;
    const size_t n = (size_t)BC * Ho * Wo * T;
    size_t idx = (size_t)blockIdx.x * blockDim.x + threadIdx.x;
    if (idx >= n) return;
    int t = (int)(idx % T); size_t r = idx / T;
    int ox = (int)(r % Wo); r /= Wo;
    int oy = (int)(r % Ho); r /= Ho;
    int bc = (int)r;
    float s = 0.f;
    for (int ky = 0; ky < k; ++ky)
        for (int kx = 0; kx < k; ++kx)
            s += in[(((size_t)bc * H + (size_t)(oy * k + ky)) * W + (size_t)(ox * k + kx)) * T + t];
    out[idx] = s * scale;
}

// ---------------------------------------------------------------------------
// Direct 2-D conv (cross-correlation), per timestep, T innermost (coalesced).
// in [B,Cin,H,W,T], w [Cout,Cin,K,K], out [B,Cout,H,W,T]
// ---------------------------------------------------------------------------
__global__ __launch_bounds__(256)
void conv_kernel(const float* __restrict__ in, const float* __restrict__ w,
                 float* __restrict__ out,
                 int B, int Cin, int Cout, int H, int W, int Ksz, int pad, int T)
{
    const size_t n = (size_t)B * Cout * H * W * T;
    size_t idx = (size_t)blockIdx.x * blockDim.x + threadIdx.x;
    if (idx >= n) return;
    int t = (int)(idx % T); size_t r = idx / T;
    int x = (int)(r % W); r /= W;
    int y = (int)(r % H); r /= H;
    int oc = (int)(r % Cout);
    int b  = (int)(r / Cout);
    float acc = 0.f;
    for (int ic = 0; ic < Cin; ++ic) {
        const float* ip = in + (((size_t)b * Cin + ic) * H) * W * T;
        const float* wp = w + ((size_t)(oc * Cin + ic)) * Ksz * Ksz;
        for (int dy = 0; dy < Ksz; ++dy) {
            int iy = y + dy - pad;
            if (iy < 0 || iy >= H) continue;
            for (int dx = 0; dx < Ksz; ++dx) {
                int ix = x + dx - pad;
                if (ix < 0 || ix >= W) continue;
                acc += ip[((size_t)iy * W + ix) * T + t] * wp[dy * Ksz + dx];
            }
        }
    }
    out[idx] = acc;
}

// ---------------------------------------------------------------------------
// Fused psp (exact IIR for alpha kernel) + spike scan with refractory decay.
// z [rows,T] -> s [rows,T]; one thread per row, sequential in T.
// ---------------------------------------------------------------------------
__global__ __launch_bounds__(256)
void iir_spike_kernel(const float* __restrict__ z, float* __restrict__ s,
                      int rows, int T)
{
    int r = blockIdx.x * blockDim.x + threadIdx.x;
    if (r >= rows) return;
    const float* zr = z + (size_t)r * T;
    float* sr = s + (size_t)r * T;
    float s1 = 0.f, s2 = 0.f, ref = 0.f;
    for (int t = 0; t < T; ++t) {
        s2 = IIR_D * (s2 + s1);          // s2[t] = d*(s2[t-1]+s1[t-1])
        s1 = IIR_D * s1 + zr[t];         // s1[t] = d*s1[t-1] + z[t]
        float u  = IIR_EC * s2;          // psp output
        float v  = u + ref - THETA;
        float sp = (v >= 0.f) ? 1.f : 0.f;
        ref = REF_D * ref - 2.f * THETA * sp;
        sr[t] = sp;
    }
}

// Final layer variant: z [4,16,T] padded channels -> d_out [4,11,T]
__global__ __launch_bounds__(64)
void iir_spike_final_kernel(const float* __restrict__ z, float* __restrict__ out, int T)
{
    int r = blockIdx.x * blockDim.x + threadIdx.x;
    if (r >= 64) return;
    int b = r >> 4, o = r & 15;
    const bool wr = (o < 11);
    const float* zr = z + (size_t)r * T;
    float* orow = out + ((size_t)(b * 11 + (wr ? o : 0))) * T;
    float s1 = 0.f, s2 = 0.f, ref = 0.f;
    for (int t = 0; t < T; ++t) {
        s2 = IIR_D * (s2 + s1);
        s1 = IIR_D * s1 + zr[t];
        float u  = IIR_EC * s2;
        float v  = u + ref - THETA;
        float sp = (v >= 0.f) ? 1.f : 0.f;
        ref = REF_D * ref - 2.f * THETA * sp;
        if (wr) orow[t] = sp;
    }
}

// ---------------------------------------------------------------------------
// Per-channel fractional temporal delay (linear interpolation, zero outside).
// in/out [B,C,HW,T]
// ---------------------------------------------------------------------------
__global__ __launch_bounds__(256)
void delay_kernel(const float* __restrict__ in, const float* __restrict__ d,
                  float* __restrict__ out, int B, int C, int HW, int T)
{
    const size_t n = (size_t)B * C * HW * T;
    size_t idx = (size_t)blockIdx.x * blockDim.x + threadIdx.x;
    if (idx >= n) return;
    int t = (int)(idx % T); size_t r = idx / T;
    size_t p = r % HW; r /= HW;
    int c = (int)(r % C);
    int b = (int)(r / C);
    float src = (float)t - d[c];
    float f = floorf(src);
    int i0 = (int)f;
    float w = src - f;
    const float* row = in + (((size_t)(b * C + c)) * HW + p) * T;
    float v0 = (i0 >= 0 && i0 < T) ? row[i0] : 0.f;
    int i1 = i0 + 1;
    float v1 = (i1 >= 0 && i1 < T) ? row[i1] : 0.f;
    out[idx] = v0 * (1.f - w) + v1 * w;
}

// ---------------------------------------------------------------------------
// f32 [B,Ci,T] -> f16 [B,Tp,Ci] transpose (zero-padded to Tp rows).
// ---------------------------------------------------------------------------
__global__ __launch_bounds__(256)
void cvt_transpose_kernel(const float* __restrict__ X, _Float16* __restrict__ Xh,
                          int B, int Ci, int T, int Tp)
{
    const size_t n = (size_t)B * Tp * Ci;
    size_t idx = (size_t)blockIdx.x * blockDim.x + threadIdx.x;
    if (idx >= n) return;
    int i = (int)(idx % Ci); size_t r = idx / Ci;
    int t = (int)(r % Tp);
    int b = (int)(r / Tp);
    float v = (t < T) ? X[((size_t)b * Ci + i) * T + t] : 0.f;
    Xh[idx] = (_Float16)v;
}

// f32 [M,K] -> f16 [Mp,K], rows >= M zero-filled.
__global__ __launch_bounds__(256)
void cvt_w_kernel(const float* __restrict__ W, _Float16* __restrict__ Wh,
                  int M, int Mp, int K)
{
    const size_t n = (size_t)Mp * K;
    size_t idx = (size_t)blockIdx.x * blockDim.x + threadIdx.x;
    if (idx >= n) return;
    int m = (int)(idx / K);
    Wh[idx] = (m < M) ? (_Float16)W[idx] : (_Float16)0.f;
}

// ---------------------------------------------------------------------------
// WMMA FC:  Z[b,m,t] = sum_k Wh[m,k] * Xh[b,t,k]
// One wave (32 lanes) computes a 16x16 (M x T) tile; K consumed 32/step via
// v_wmma_f32_16x16x32_f16.  gfx1250 VGPR layouts per CDNA5 ISA 7.12.2:
//   A 16x32: lanes 0-15 & 16-31 hold M=0..15; lane-group selects K-halves.
//   B 32x16: lanes 0-15 -> K=0..7,16..23, lanes 16-31 -> K=8..15,24..31, N = lane&15.
//   C/D: VGPR r -> M=r (lanes 0-15) / M=r+8 (lanes 16-31), N = lane&15.
// ---------------------------------------------------------------------------
__global__ __launch_bounds__(32)
void wmma_fc_kernel(const _Float16* __restrict__ Wh,  // [M,K] row-major
                    const _Float16* __restrict__ Xh,  // [B,Tp,K] row-major
                    float* __restrict__ Z,            // [B,M,T]
                    int M, int K, int T, int Tp)
{
    const int m0 = blockIdx.x * 16;
    const int t0 = blockIdx.y * 16;
    const int b  = blockIdx.z;
    const int lane = threadIdx.x;
    const int g  = lane >> 4;       // lane half-group
    const int li = lane & 15;

    const _Float16* Xb = Xh + (size_t)b * Tp * K;
    const size_t arow = (size_t)(m0 + li) * K;   // A row = M
    const size_t brow = (size_t)(t0 + li) * K;   // B column = t (row in [T,K] layout)

    union { v16h h; unsigned int u[8]; } Af, Bf;
    v8f acc = {0.f, 0.f, 0.f, 0.f, 0.f, 0.f, 0.f, 0.f};

    for (int k0 = 0; k0 < K; k0 += 32) {
        // A fragment: VGPR v<4 -> K = g*8 + 2v ; v>=4 -> K = 16 + g*8 + 2(v-4)
#pragma unroll
        for (int v = 0; v < 8; ++v) {
            int kA = (v < 4) ? (g * 8 + 2 * v) : (16 + g * 8 + 2 * (v - 4));
            Af.u[v] = *reinterpret_cast<const unsigned int*>(Wh + arow + k0 + kA);
        }
        // B fragment: VGPR v<4 -> K = g*8 + 2v ; v>=4 -> K = 16 + g*8 + 2(v-4)
#pragma unroll
        for (int v = 0; v < 8; ++v) {
            int kB = (v < 4) ? (g * 8 + 2 * v) : (16 + g * 8 + 2 * (v - 4));
            Bf.u[v] = *reinterpret_cast<const unsigned int*>(Xb + brow + k0 + kB);
        }
        acc = __builtin_amdgcn_wmma_f32_16x16x32_f16(
            /*neg_a=*/false, Af.h, /*neg_b=*/false, Bf.h,
            /*c_mod=*/(short)0, acc, /*reuse_a=*/false, /*reuse_b=*/false);
    }

    const int t = t0 + li;
    if (t < T) {
#pragma unroll
        for (int r = 0; r < 8; ++r) {
            int m = m0 + r + 8 * g;
            Z[((size_t)b * M + m) * T + t] = acc[r];
        }
    }
}

// ---------------------------------------------------------------------------
// Host-side orchestration
// ---------------------------------------------------------------------------
extern "C" void kernel_launch(void* const* d_in, const int* in_sizes, int n_in,
                              void* d_out, int out_size, void* d_ws, size_t ws_size,
                              hipStream_t stream)
{
    (void)in_sizes; (void)n_in; (void)out_size; (void)ws_size;

    const float* x0  = (const float*)d_in[0];  // [4,2,128,128,300]
    const float* w1  = (const float*)d_in[1];  // [16,2,5,5]
    const float* w2  = (const float*)d_in[2];  // [32,16,3,3]
    const float* fc1 = (const float*)d_in[3];  // [512,2048]
    const float* fc2 = (const float*)d_in[4];  // [11,512]
    const float* dl1 = (const float*)d_in[5];  // [16]
    const float* dl2 = (const float*)d_in[6];  // [32]
    const float* dl3 = (const float*)d_in[7];  // [512]
    float* out = (float*)d_out;                // [4,11,300]

    const int B = 4, T = 300, Tp = 304;        // Tp = 19*16 for WMMA tiling

    // workspace layout: 3 rotating f32 regions (sized for biggest tensor) + f16 staging
    const size_t BIGF = (size_t)4 * 16 * 32 * 32 * 300;   // 19,660,800 floats
    float* bufA = (float*)d_ws;
    float* bufB = bufA + BIGF;
    float* bufC = bufB + BIGF;
    _Float16* Xh1 = (_Float16*)(bufC + BIGF);             // [4,304,2048]
    _Float16* W1h = Xh1 + (size_t)B * Tp * 2048;          // [512,2048]
    _Float16* Xh2 = W1h + (size_t)512 * 2048;             // [4,304,512]
    _Float16* W2h = Xh2 + (size_t)B * Tp * 512;           // [16,512]

    const int BK = 256;
    size_t n;

    // L1: pool4 (commuted before psp) -> IIR+spike       [4,2,32,32,T]
    n = (size_t)8 * 32 * 32 * T;
    pool_kernel<<<cdiv(n, BK), BK, 0, stream>>>(x0, bufA, 8, 128, 128, 4, T, 1.1f * THETA / 16.f);
    iir_spike_kernel<<<cdiv(8192, BK), BK, 0, stream>>>(bufA, bufB, 8192, T);

    // L2: conv1 5x5 (2->16) -> IIR+spike -> delay1       [4,16,32,32,T]
    n = (size_t)B * 16 * 32 * 32 * T;
    conv_kernel<<<cdiv(n, BK), BK, 0, stream>>>(bufB, w1, bufA, B, 2, 16, 32, 32, 5, 2, T);
    iir_spike_kernel<<<cdiv(65536, BK), BK, 0, stream>>>(bufA, bufB, 65536, T);
    delay_kernel<<<cdiv(n, BK), BK, 0, stream>>>(bufB, dl1, bufC, B, 16, 1024, T);

    // L3: pool2 -> IIR+spike                             [4,16,16,16,T]
    n = (size_t)64 * 16 * 16 * T;
    pool_kernel<<<cdiv(n, BK), BK, 0, stream>>>(bufC, bufA, 64, 32, 32, 2, T, 1.1f * THETA / 4.f);
    iir_spike_kernel<<<cdiv(16384, BK), BK, 0, stream>>>(bufA, bufB, 16384, T);

    // L4: conv2 3x3 (16->32) -> IIR+spike -> delay2      [4,32,16,16,T]
    n = (size_t)B * 32 * 16 * 16 * T;
    conv_kernel<<<cdiv(n, BK), BK, 0, stream>>>(bufB, w2, bufA, B, 16, 32, 16, 16, 3, 1, T);
    iir_spike_kernel<<<cdiv(32768, BK), BK, 0, stream>>>(bufA, bufB, 32768, T);
    delay_kernel<<<cdiv(n, BK), BK, 0, stream>>>(bufB, dl2, bufC, B, 32, 256, T);

    // L5: pool2 -> IIR+spike                             [4,32,8,8,T] == [4,2048,T]
    n = (size_t)128 * 8 * 8 * T;
    pool_kernel<<<cdiv(n, BK), BK, 0, stream>>>(bufC, bufA, 128, 16, 16, 2, T, 1.1f * THETA / 4.f);
    iir_spike_kernel<<<cdiv(8192, BK), BK, 0, stream>>>(bufA, bufB, 8192, T);

    // L6: fc1 (512x2048) via WMMA -> IIR+spike -> delay3
    n = (size_t)B * Tp * 2048;
    cvt_transpose_kernel<<<cdiv(n, BK), BK, 0, stream>>>(bufB, Xh1, B, 2048, T, Tp);
    n = (size_t)512 * 2048;
    cvt_w_kernel<<<cdiv(n, BK), BK, 0, stream>>>(fc1, W1h, 512, 512, 2048);
    {
        dim3 g(512 / 16, Tp / 16, B);
        wmma_fc_kernel<<<g, 32, 0, stream>>>(W1h, Xh1, bufA, 512, 2048, T, Tp);
    }
    iir_spike_kernel<<<cdiv(2048, BK), BK, 0, stream>>>(bufA, bufB, 2048, T);
    n = (size_t)B * 512 * T;
    delay_kernel<<<cdiv(n, BK), BK, 0, stream>>>(bufB, dl3, bufC, B, 512, 1, T);

    // L7: fc2 (11x512, padded to 16) via WMMA -> final IIR+spike -> d_out
    n = (size_t)B * Tp * 512;
    cvt_transpose_kernel<<<cdiv(n, BK), BK, 0, stream>>>(bufC, Xh2, B, 512, T, Tp);
    n = (size_t)16 * 512;
    cvt_w_kernel<<<cdiv(n, BK), BK, 0, stream>>>(fc2, W2h, 11, 16, 512);
    {
        dim3 g(1, Tp / 16, B);
        wmma_fc_kernel<<<g, 32, 0, stream>>>(W2h, Xh2, bufA, 16, 512, T, Tp);
    }
    iir_spike_final_kernel<<<1, 64, 0, stream>>>(bufA, out, T);
}
